// Attention_54589034332712
// MI455X (gfx1250) — matrix operand order repaired and checked by reading, not compiled, not run on hardware
//
#include <hip/hip_runtime.h>

// MI455X (gfx1250) wave32 WMMA attention block.
// WS layout (f16 elems): Q[4M] | K[4M] | Vt[4M] | Aout[4M] | HS[4M] | WqkvT[3M] | WprojT[1M]
//                      = 48 MB total.

typedef _Float16 f16;
typedef __attribute__((ext_vector_type(16))) _Float16 v16h;
typedef __attribute__((ext_vector_type(8)))  _Float16 v8h;
typedef __attribute__((ext_vector_type(4)))  _Float16 v4h;
typedef __attribute__((ext_vector_type(8)))  float    v8f;

#define WMMA_F32_F16(a, b, c) \
    __builtin_amdgcn_wmma_f32_16x16x32_f16(false, (a), false, (b), (short)0, (c), false, false)

static constexpr int Bb = 4, Ss = 1024, Ee = 1024, Hh = 16, Dd = 64;
static constexpr int HEAD_ELEMS = Bb * Hh * Ss * Dd;      // 4 Mi
static constexpr float kBigPos = 3.4028234663852886e38f;  // -finfo(f32).min

// ---------------------------------------------------------------------------
// Prep 1: hidden_states f32 -> f16, elementwise (4 elems/thread).
// ---------------------------------------------------------------------------
__global__ __launch_bounds__(256) void cvt_hs_kernel(const float* __restrict__ in,
                                                     f16* __restrict__ out)
{
    const int i = (blockIdx.x * 256 + threadIdx.x) * 4;
    float4 v = *(const float4*)(in + i);
    v4h o;
    o[0] = (f16)v.x; o[1] = (f16)v.y; o[2] = (f16)v.z; o[3] = (f16)v.w;
    *(v4h*)(out + i) = o;
}

// ---------------------------------------------------------------------------
// Prep 2: transpose + convert weights: in (R x C, f32) -> out (C x R, f16).
// LDS 32x33 tile, both global sides coalesced. Block (32,8), grid (C/32, R/32).
// ---------------------------------------------------------------------------
__global__ __launch_bounds__(256) void transpose_cvt_kernel(const float* __restrict__ in,
                                                            f16* __restrict__ out,
                                                            int R, int C)
{
    __shared__ f16 tile[32][33];
    const int tx = threadIdx.x, ty = threadIdx.y;
    const int c0 = blockIdx.x * 32, r0 = blockIdx.y * 32;
#pragma unroll
    for (int j = 0; j < 4; ++j)
        tile[ty + j * 8][tx] = (f16)in[(size_t)(r0 + ty + j * 8) * C + c0 + tx];
    __syncthreads();
#pragma unroll
    for (int j = 0; j < 4; ++j)
        out[(size_t)(c0 + ty + j * 8) * R + r0 + tx] = tile[tx][ty + j * 8];
}

// ---------------------------------------------------------------------------
// Shared fragment loader (16-bit A fragment, contiguous k = 16*hlf + s, same order as B).
// ---------------------------------------------------------------------------
__device__ __forceinline__ v16h load_a_frag(const f16* __restrict__ arow, int kk, int hlf)
{
    const int ka = kk * 32 + hlf * 16;
    v8h lo = *(const v8h*)(arow + ka);
    v8h hi = *(const v8h*)(arow + ka + 8);
    v16h af;
#pragma unroll
    for (int e = 0; e < 8; ++e) { af[e] = lo[e]; af[8 + e] = hi[e]; }
    return af;
}

// Fully-unrolled 16Mx64N GEMM body over K=1024 (32 steps of 32). Straight-line
// code: every fragment is a fresh SSA value, letting the scheduler hoist loads
// across earlier WMMAs with no loop-carried register copies.
template <int NSTRIDE>
__device__ __forceinline__ void gemm_16x64_body(const f16* __restrict__ arow,
                                                const f16* __restrict__ wbase,
                                                int hlf, v8f acc[4])
{
#pragma unroll
    for (int kk = 0; kk < 32; ++kk) {
        v16h af = load_a_frag(arow, kk, hlf);
        v16h bf[4];
#pragma unroll
        for (int nt = 0; nt < 4; ++nt)
            bf[nt] = *(const v16h*)(wbase + (size_t)nt * NSTRIDE + kk * 32);
#pragma unroll
        for (int nt = 0; nt < 4; ++nt)
            acc[nt] = WMMA_F32_F16(af, bf[nt], acc[nt]);
    }
}

// ---------------------------------------------------------------------------
// Kernel 1: QKV projection. C(4096x3072) = HS(4096x1024,f16) x WqkvT(f16,N-major) + b
// Writes Q (scaled 1/8) and K in (B,H,S,D); V transposed (B,H,D,S).
// ---------------------------------------------------------------------------
__global__ __launch_bounds__(256) void qkv_kernel(
    const f16* __restrict__ hsh, const f16* __restrict__ wt,
    const float* __restrict__ bias,
    f16* __restrict__ qw, f16* __restrict__ kw, f16* __restrict__ vtw)
{
    const int lane = threadIdx.x & 31;
    const int wave = threadIdx.x >> 5;
    const int l15  = lane & 15;
    const int hlf  = lane >> 4;

    const int wt_id = blockIdx.x * 8 + wave;   // 12288 wave tiles
    const int mtile = wt_id / 48;              // 256 row tiles of 16
    const int ngrp  = wt_id % 48;              // 48 col groups of 64

    const f16* arow  = hsh + (size_t)(mtile * 16 + l15) * Ee;
    const f16* wbase = wt + (size_t)(ngrp * 64 + l15) * 1024 + hlf * 16;

    v8f acc[4] = {};
    gemm_16x64_body<16 * 1024>(arow, wbase, hlf, acc);

#pragma unroll
    for (int nt = 0; nt < 4; ++nt) {
        const int col   = ngrp * 64 + nt * 16 + l15;   // 0..3071
        const int which = col >> 10;                   // 0=Q 1=K 2=V (wave-uniform)
        const int e     = col & 1023;
        const int h     = e >> 6;
        const int d     = e & 63;
        const float bv  = bias[col];
#pragma unroll
        for (int r = 0; r < 8; ++r) {
            const int row = mtile * 16 + r + hlf * 8;  // 0..4095
            const int b   = row >> 10;
            const int s   = row & 1023;
            const float v = acc[nt][r] + bv;
            if (which == 0)
                qw[(((b * Hh + h) * Ss + s) << 6) + d] = (f16)(v * 0.125f); // fold 1/sqrt(64)
            else if (which == 1)
                kw[(((b * Hh + h) * Ss + s) << 6) + d] = (f16)v;
            else
                vtw[((b * Hh + h) * Dd + d) * Ss + s] = (f16)v;             // transposed
        }
    }
}

// ---------------------------------------------------------------------------
// Kernel 2: per-head attention, flash-style online softmax, 16 queries/wave,
// 32-key tiles. Reproduces the reference's "w*mask - finfo.min*(1-mask)"
// (i.e. +FLT_MAX at future positions) exactly.
// ---------------------------------------------------------------------------
__global__ __launch_bounds__(128) void attn_kernel(
    const f16* __restrict__ qw, const f16* __restrict__ kw,
    const f16* __restrict__ vtw, f16* __restrict__ aout)
{
    __shared__ __align__(32) f16 plds[4][16 * 32];   // per-wave P transpose buffer

    const int lane = threadIdx.x & 31;
    const int wave = threadIdx.x >> 5;
    const int l15  = lane & 15;
    const int hlf  = lane >> 4;

    const int idx = blockIdx.x * 4 + wave;  // 4096 wave tiles
    const int qt  = idx & 63;
    const int bh  = idx >> 6;               // 0..63
    const int b   = bh >> 4;
    const int h   = bh & 15;

    const f16* Q  = qw  + (size_t)bh * Ss * Dd;
    const f16* K  = kw  + (size_t)bh * Ss * Dd;
    const f16* Vt = vtw + (size_t)bh * Ss * Dd;
    const int  q0 = qt * 16;

    v16h qf[2];
#pragma unroll
    for (int kc = 0; kc < 2; ++kc) {
        const f16* qr = Q + (q0 + l15) * Dd + kc * 32 + hlf * 16;
        v8h lo = *(const v8h*)qr;
        v8h hi = *(const v8h*)(qr + 8);
#pragma unroll
        for (int e = 0; e < 8; ++e) { qf[kc][e] = lo[e]; qf[kc][8 + e] = hi[e]; }
    }

    v8f   o[4] = {};
    float mi[8], li[8];
#pragma unroll
    for (int r = 0; r < 8; ++r) { mi[r] = -kBigPos; li[r] = 0.0f; }

    f16* pl = plds[wave];

    for (int kt = 0; kt < 32; ++kt) {
        // ---- scores: S(16x32) = Q(16x64) x K^T(64x32), 4 WMMA ----
        v8f sc[2] = {};
#pragma unroll
        for (int nt = 0; nt < 2; ++nt) {
            const int key = kt * 32 + nt * 16 + l15;
#pragma unroll
            for (int kc = 0; kc < 2; ++kc) {
                v16h kf = *(const v16h*)(K + key * Dd + kc * 32 + hlf * 16);
                sc[nt] = WMMA_F32_F16(qf[kc], kf, sc[nt]);
            }
        }

        // ---- issue V-fragment loads early: overlap with softmax VALU work ----
        v16h vf[4];
#pragma unroll
        for (int dt = 0; dt < 4; ++dt)
            vf[dt] = *(const v16h*)(Vt + (dt * 16 + l15) * Ss + kt * 32 + hlf * 16);

        // ---- mask + online softmax (row r+8*hlf lives in this 16-lane half) ----
#pragma unroll
        for (int r = 0; r < 8; ++r) {
            const int qrow = q0 + r + hlf * 8;
            const int key0 = kt * 32 + l15;
            const int key1 = key0 + 16;
            float s0 = (key0 > qrow) ? kBigPos : sc[0][r];  // faithful +FLT_MAX trick
            float s1 = (key1 > qrow) ? kBigPos : sc[1][r];

            float vmax = fmaxf(s0, s1);
            vmax = fmaxf(vmax, __shfl_xor(vmax, 1, 32));
            vmax = fmaxf(vmax, __shfl_xor(vmax, 2, 32));
            vmax = fmaxf(vmax, __shfl_xor(vmax, 4, 32));
            vmax = fmaxf(vmax, __shfl_xor(vmax, 8, 32));
            const float mnew  = fmaxf(mi[r], vmax);
            const float scale = __expf(mi[r] - mnew);
            const float p0 = __expf(s0 - mnew);
            const float p1 = __expf(s1 - mnew);
            float psum = p0 + p1;
            psum += __shfl_xor(psum, 1, 32);
            psum += __shfl_xor(psum, 2, 32);
            psum += __shfl_xor(psum, 4, 32);
            psum += __shfl_xor(psum, 8, 32);
            li[r] = li[r] * scale + psum;
            mi[r] = mnew;
#pragma unroll
            for (int dt = 0; dt < 4; ++dt) o[dt][r] *= scale;

            const int mrow = r + hlf * 8;
            pl[mrow * 32 + l15]      = (f16)p0;   // C-layout -> row-major LDS
            pl[mrow * 32 + 16 + l15] = (f16)p1;
        }

        // Within-wave DS stores must land before the transposed reload.
        asm volatile("s_wait_dscnt 0x0" ::: "memory");

        // ---- reload P as an A-fragment (16x32) ----
        v16h pf;
        {
            const f16* pr = pl + l15 * 32 + hlf * 16;
            v8h lo = *(const v8h*)pr;
            v8h hi = *(const v8h*)(pr + 8);
#pragma unroll
            for (int e = 0; e < 8; ++e) { pf[e] = lo[e]; pf[8 + e] = hi[e]; }
        }
        asm volatile("s_wait_dscnt 0x0" ::: "memory");

        // ---- O(16x64) += P(16x32) x V(32x64): 4 WMMA with preloaded V ----
#pragma unroll
        for (int dt = 0; dt < 4; ++dt)
            o[dt] = WMMA_F32_F16(pf, vf[dt], o[dt]);
    }

    // ---- normalize and merge heads: aout[(b*S+s)][h*64+d], f16 ----
#pragma unroll
    for (int dt = 0; dt < 4; ++dt) {
        const int dim = dt * 16 + l15;
#pragma unroll
        for (int r = 0; r < 8; ++r) {
            const int srow = q0 + r + hlf * 8;
            const float val = o[dt][r] / li[r];
            aout[((size_t)(b * Ss + srow)) * Ee + h * Dd + dim] = (f16)val;
        }
    }
}

// ---------------------------------------------------------------------------
// Kernel 3: output projection. out(4096x1024) = A(4096x1024,f16) x WprojT + bp
// ---------------------------------------------------------------------------
__global__ __launch_bounds__(256) void proj_kernel(
    const f16* __restrict__ a, const f16* __restrict__ wt,
    const float* __restrict__ bias, float* __restrict__ out)
{
    const int lane = threadIdx.x & 31;
    const int wave = threadIdx.x >> 5;
    const int l15  = lane & 15;
    const int hlf  = lane >> 4;

    const int wt_id = blockIdx.x * 8 + wave;  // 4096 wave tiles
    const int mtile = wt_id >> 4;             // 256 row tiles
    const int ngrp  = wt_id & 15;             // 16 col groups of 64

    const f16* arow  = a + (size_t)(mtile * 16 + l15) * Ee;
    const f16* wbase = wt + (size_t)(ngrp * 64 + l15) * Ee + hlf * 16;

    v8f acc[4] = {};
    gemm_16x64_body<16 * Ee>(arow, wbase, hlf, acc);

#pragma unroll
    for (int nt = 0; nt < 4; ++nt) {
        const int col  = ngrp * 64 + nt * 16 + l15;
        const float bv = bias[col];
#pragma unroll
        for (int r = 0; r < 8; ++r) {
            const int row = mtile * 16 + r + hlf * 8;
            out[(size_t)row * Ee + col] = acc[nt][r] + bv;
        }
    }
}

extern "C" void kernel_launch(void* const* d_in, const int* in_sizes, int n_in,
                              void* d_out, int out_size, void* d_ws, size_t ws_size,
                              hipStream_t stream) {
    const float* hs    = (const float*)d_in[0];
    const float* wqkv  = (const float*)d_in[1];
    const float* bqkv  = (const float*)d_in[2];
    const float* wproj = (const float*)d_in[3];
    const float* bproj = (const float*)d_in[4];
    // d_in[5] = attn_mask (tril) — reproduced analytically in-kernel.

    f16* qw    = (f16*)d_ws;
    f16* kw    = qw    + HEAD_ELEMS;           //  8 MB
    f16* vtw   = kw    + HEAD_ELEMS;           // 16 MB
    f16* aout  = vtw   + HEAD_ELEMS;           // 24 MB
    f16* hsh   = aout  + HEAD_ELEMS;           // 32 MB
    f16* wqkvT = hsh   + HEAD_ELEMS;           // 40 MB
    f16* wprjT = wqkvT + (size_t)3072 * 1024;  // 46 MB (total 48 MB)

    cvt_hs_kernel<<<4096, 256, 0, stream>>>(hs, hsh);
    transpose_cvt_kernel<<<dim3(96, 32), dim3(32, 8), 0, stream>>>(wqkv, wqkvT, 1024, 3072);
    transpose_cvt_kernel<<<dim3(32, 32), dim3(32, 8), 0, stream>>>(wproj, wprjT, 1024, 1024);

    qkv_kernel <<<1536, 256, 0, stream>>>(hsh, wqkvT, bqkv, qw, kw, vtw);
    attn_kernel<<<1024, 128, 0, stream>>>(qw, kw, vtw, aout);
    proj_kernel<<< 512, 256, 0, stream>>>(aout, wprjT, bproj, (float*)d_out);
}
